// UIA_M_Model_51642686767387
// MI455X (gfx1250) — matrix operand order repaired and checked by reading, not compiled, not run on hardware
//
#include <hip/hip_runtime.h>

// ---------------------------------------------------------------------------
// MI455X (gfx1250). GEMMs -> v_wmma_f32_16x16x32_bf16 (wave32, f32 accum).
// A-operands stored bf16 in memory (halves dominant HBM traffic vs f32).
// A tiles staged LDS-side by the Tensor Data Mover (tensor_load_to_lds with
// LDS row padding), double-buffered via TENSORcnt. Block tile 128x64, 8 waves.
// ---------------------------------------------------------------------------

typedef __attribute__((ext_vector_type(16))) __bf16 v16bf;
typedef __attribute__((ext_vector_type(8)))  float  v8f;
typedef __attribute__((ext_vector_type(4)))  unsigned int u32x4;
typedef __attribute__((ext_vector_type(8)))  int          i32x8;
typedef __attribute__((ext_vector_type(4)))  int          i32x4;

#define BATCH 32768L

__device__ __forceinline__ float gelu_f(float x) {
    return 0.5f * x * (1.0f + erff(x * 0.70710678118f));
}
__device__ __forceinline__ float gelu_g(float x) {
    return 0.5f * (1.0f + erff(x * 0.70710678118f))
         + x * 0.3989422804f * expf(-0.5f * x * x);
}

// ---- TDM: DMA a 128x32 bf16 tile (row stride lda elems) into LDS with a
// 16B pad after every 64B row (-> 40-half LDS row stride). ISA 08 §8.3/8.4.
// This toolchain exposes the 6-arg builtin: (g0, g1, g2, g3, g_extra, cpol).
__device__ __forceinline__ void tdm_load_tile_bf16(
    unsigned lds_addr, const __bf16* gaddr, int lda)
{
    unsigned long ga = (unsigned long)gaddr;
    u32x4 g0 = {
        1u,                                               // count=1, load, user
        lds_addr,                                         // LDS dest (bytes)
        (unsigned)(ga & 0xFFFFFFFFu),                     // global_addr[31:0]
        (unsigned)((ga >> 32) & 0x1FFFFFFu) | (2u << 30)  // ga[56:32] | type=2
    };
    i32x8 g1 = {
        (int)((1u << 16) | (1u << 20) | (3u << 22) | (3u << 25)),
        //    data_size=2B | pad_en | pad_interval=16dw | pad_amount=4dw
        (int)(32u  << 16),   // tensor_dim0 = 32
        (int)(128u << 16),   // tensor_dim1 = 128
        (int)(32u  << 16),   // tile_dim0   = 32
        128,                 // tile_dim1   = 128
        lda,                 // tensor_dim0_stride (elements)
        0, 0
    };
    i32x4 z4 = { 0, 0, 0, 0 };
    i32x8 z8 = { 0, 0, 0, 0, 0, 0, 0, 0 };
    __builtin_amdgcn_tensor_load_to_lds(g0, g1, z4, z4, z8, 0);
}

// ---------------------------------------------------------------------------
// C[M,N] = alpha * (A[M,K](bf16) @ op(W)(f32->bf16)) + bias; optional ReLU.
//  TRANSB=false: W row-major KxN (ldb=N). TRANSB=true: W row-major NxK (ldb=K).
// Outputs: C (f32, nullable) and/or Cb (bf16, nullable).
// M multiple of 128, K multiple of 32, N arbitrary (predicated tail tile).
// ---------------------------------------------------------------------------
template <bool TRANSB>
__global__ __launch_bounds__(256) void gemm_wmma(
    const __bf16* __restrict__ A, const float* __restrict__ W,
    const float* __restrict__ bias, float* __restrict__ C,
    __bf16* __restrict__ Cb,
    int N, int K, int lda, int ldb, int ldc, float alpha, int act)
{
    constexpr int BM = 128, BN = 64, KT = 32, AS = 40;
    __shared__ __bf16 lA[2][BM * AS];   // double-buffered TDM destination
    __shared__ __bf16 lB[BN * AS];

    const int tid  = threadIdx.x;
    const int lane = tid & 31;
    const int wid  = tid >> 5;
    const int wm   = wid & 3;
    const int wn   = wid >> 2;
    const long m0  = (long)blockIdx.y * BM;
    const int  n0  = blockIdx.x * BN;
    const bool fullN = (n0 + BN) <= N;

    v8f acc[2][2] = {};
    const int ksteps = K / KT;
    const __bf16* arow = A + m0 * (long)lda;

    // prime: DMA tile 0 (wave 0 issues; TDM ignores EXEC, one op per wave)
    if (wid == 0)
        tdm_load_tile_bf16((unsigned)(unsigned long)&lA[0][0], arow, lda);

    for (int kt = 0; kt < ksteps; ++kt) {
        const int k0 = kt * KT;
        const bool more = (kt + 1) < ksteps;

        // ---- issue DMA for next A tile while we stage B / compute ----
        if (wid == 0 && more)
            tdm_load_tile_bf16((unsigned)(unsigned long)&lA[(kt + 1) & 1][0],
                               arow + (k0 + KT), lda);

        // ---- stage B tile into LDS as [n][k] (transpose + cvt in VALU) ----
        if (!TRANSB) {
            const int k  = tid >> 3;
            const int nn = (tid & 7) * 8;
            const float* gw = W + (long)(k0 + k) * ldb + n0 + nn;
            if (fullN) {            // uniform fast path: no exec juggling
                float4 f0 = *reinterpret_cast<const float4*>(gw);
                float4 f1 = *reinterpret_cast<const float4*>(gw + 4);
                lB[(nn + 0) * AS + k] = (__bf16)f0.x;
                lB[(nn + 1) * AS + k] = (__bf16)f0.y;
                lB[(nn + 2) * AS + k] = (__bf16)f0.z;
                lB[(nn + 3) * AS + k] = (__bf16)f0.w;
                lB[(nn + 4) * AS + k] = (__bf16)f1.x;
                lB[(nn + 5) * AS + k] = (__bf16)f1.y;
                lB[(nn + 6) * AS + k] = (__bf16)f1.z;
                lB[(nn + 7) * AS + k] = (__bf16)f1.w;
            } else {                // classifier tail tile only
#pragma unroll
                for (int v = 0; v < 8; ++v) {
                    float f = (n0 + nn + v < N) ? gw[v] : 0.0f;
                    lB[(nn + v) * AS + k] = (__bf16)f;
                }
            }
        } else {
            const int n  = tid >> 2;
            const int kk = (tid & 3) * 8;
            const float* gw = W + (long)(n0 + n) * ldb + k0 + kk;
            if (n0 + n < N) {       // per-thread uniform predicate
                float4 f0 = *reinterpret_cast<const float4*>(gw);
                float4 f1 = *reinterpret_cast<const float4*>(gw + 4);
                __bf16* d = &lB[n * AS + kk];
                d[0] = (__bf16)f0.x; d[1] = (__bf16)f0.y;
                d[2] = (__bf16)f0.z; d[3] = (__bf16)f0.w;
                d[4] = (__bf16)f1.x; d[5] = (__bf16)f1.y;
                d[6] = (__bf16)f1.z; d[7] = (__bf16)f1.w;
            } else {
                __bf16* d = &lB[n * AS + kk];
#pragma unroll
                for (int v = 0; v < 8; ++v) d[v] = (__bf16)0.0f;
            }
        }

        // ---- wave 0: tile kt has landed when TENSORcnt <= #in-flight-1 ----
        if (wid == 0) {
            if (more) __builtin_amdgcn_s_wait_tensorcnt(1);
            else      __builtin_amdgcn_s_wait_tensorcnt(0);
        }
        __syncthreads();

        // ---- fragments per ISA 7.12.2 (wave32 16-bit layouts) ----
        const __bf16* la = &lA[kt & 1][0];
        const int ksel = (lane >> 4) << 3;
        const int kbb  = (lane >> 4) << 3;
        v16bf af[2], bf[2];
#pragma unroll
        for (int i = 0; i < 2; ++i) {
            const __bf16* pr = &la[(wm * 32 + i * 16 + (lane & 15)) * AS];
#pragma unroll
            for (int e = 0; e < 8; ++e) {
                af[i][e]     = pr[ksel + e];
                af[i][8 + e] = pr[16 + ksel + e];
            }
        }
#pragma unroll
        for (int j = 0; j < 2; ++j) {
            const __bf16* pc = &lB[(wn * 32 + j * 16 + (lane & 15)) * AS + kbb];
#pragma unroll
            for (int e = 0; e < 16; ++e) bf[j][e] = pc[(e & 7) + ((e >> 3) << 4)];
        }
#pragma unroll
        for (int i = 0; i < 2; ++i)
#pragma unroll
            for (int j = 0; j < 2; ++j)
                acc[i][j] = __builtin_amdgcn_wmma_f32_16x16x32_bf16(
                    false, af[i], false, bf[j], (short)0, acc[i][j],
                    false, false);
        __syncthreads();
    }

    // ---- epilogue ----
#pragma unroll
    for (int i = 0; i < 2; ++i) {
#pragma unroll
        for (int j = 0; j < 2; ++j) {
            const int col = n0 + wn * 32 + j * 16 + (lane & 15);
            if (col < N) {
#pragma unroll
                for (int r = 0; r < 8; ++r) {
                    const long row = m0 + wm * 32 + i * 16 + ((lane >> 4) << 3) + r;
                    float v = alpha * acc[i][j][r];
                    if (bias) v += bias[col];
                    if (act == 1) v = fmaxf(v, 0.0f);
                    if (C)  C[row * (long)ldc + col]  = v;
                    if (Cb) Cb[row * (long)ldc + col] = (__bf16)v;
                }
            }
        }
    }
}

// ---- wave-per-row LayerNorm + exact GELU; bf16 out (+ optional f32 copy) ----
__global__ __launch_bounds__(256) void ln_gelu_rows(
    const float* __restrict__ X, __bf16* __restrict__ Yh,
    float* __restrict__ Yf,
    const float* __restrict__ g, const float* __restrict__ beta, int N)
{
    const int lane = threadIdx.x & 31;
    const long row = (long)blockIdx.x * 8 + (threadIdx.x >> 5);
    const float* x = X + row * N;
    float s = 0.f, s2 = 0.f;
    for (int c = lane; c < N; c += 32) { float v = x[c]; s += v; s2 += v * v; }
#pragma unroll
    for (int m = 16; m; m >>= 1) {
        s  += __shfl_xor(s,  m, 32);
        s2 += __shfl_xor(s2, m, 32);
    }
    const float mean = s / N;
    const float rstd = rsqrtf(fmaxf(s2 / N - mean * mean, 0.f) + 1e-5f);
    for (int c = lane; c < N; c += 32) {
        float v = gelu_f((x[c] - mean) * rstd * g[c] + beta[c]);
        Yh[row * N + c] = (__bf16)v;
        if (Yf) Yf[row * N + c] = v;
    }
}

// ---- small elementwise / utility kernels ----
__global__ void k_cvt(const float* __restrict__ s, __bf16* __restrict__ d, long n) {
    long i = blockIdx.x * 256L + threadIdx.x;
    if (i < n) d[i] = (__bf16)s[i];
}
__global__ void k_pool(const float* __restrict__ x, __bf16* __restrict__ p) {
    long i = blockIdx.x * 256L + threadIdx.x;          // B*512 threads
    long b = i >> 9; long c = i & 511;
    p[i] = (__bf16)(0.5f * (x[b * 1024 + 2 * c] + x[b * 1024 + 2 * c + 1]));
}
__global__ void k_gelu(const float* __restrict__ s, __bf16* __restrict__ d, long n) {
    long i = blockIdx.x * 256L + threadIdx.x;
    if (i < n) d[i] = (__bf16)gelu_f(s[i]);
}
__global__ void k_w3grad(const float* __restrict__ a, const float* __restrict__ w3s,
                         __bf16* __restrict__ d, long n) {
    long i = blockIdx.x * 256L + threadIdx.x;          // n = B*128
    if (i < n) d[i] = (__bf16)(w3s[i & 127] * gelu_g(a[i]));
}
__global__ void k_mulgrad(const float* __restrict__ dh, const float* __restrict__ a,
                          __bf16* __restrict__ d, long n) {
    long i = blockIdx.x * 256L + threadIdx.x;
    if (i < n) d[i] = (__bf16)(dh[i] * gelu_g(a[i]));
}
__global__ void k_belief(const float* __restrict__ h, const float* __restrict__ dz,
                         float* __restrict__ bel, __bf16* __restrict__ belh, long n) {
    long i = blockIdx.x * 256L + threadIdx.x;          // n = B*256
    if (i < n) { float v = h[i] + 0.1f * dz[i]; bel[i] = v; belh[i] = (__bf16)v; }
}
__global__ void k_concat_hf(__bf16* __restrict__ d, const __bf16* __restrict__ s1,
                            const float* __restrict__ s2) {   // [bf16 | f32] -> bf16
    long i = blockIdx.x * 256L + threadIdx.x;          // B*512 threads
    long b = i >> 9; long c = i & 511;                 // uniform per block
    d[i] = (c < 256) ? s1[b * 256 + c] : (__bf16)s2[b * 256 + (c - 256)];
}
__global__ void k_concat_fh(__bf16* __restrict__ d, const float* __restrict__ s1,
                            const __bf16* __restrict__ s2) {  // [f32 | bf16] -> bf16
    long i = blockIdx.x * 256L + threadIdx.x;
    long b = i >> 9; long c = i & 511;
    d[i] = (c < 256) ? (__bf16)s1[b * 256 + c] : s2[b * 256 + (c - 256)];
}
__global__ void k_cat3(__bf16* __restrict__ d, const float* __restrict__ s0,
                       const float* __restrict__ s1, const float* __restrict__ s2) {
    long i = blockIdx.x * 256L + threadIdx.x;          // B*192 threads
    long b = i / 192; long c = i - b * 192;
    float v;
    if (c < 64)       v = s0[b * 256 + c];
    else if (c < 128) v = s1[b * 128 + (c - 64)];
    else              v = s2[b * 64 + (c - 128)];
    d[i] = (__bf16)v;
}
__global__ void k_w3s(const float* __restrict__ hW3, float* __restrict__ w3s) {
    int j = threadIdx.x;                                // 128 threads
    float s = 0.f;
    for (int k = 0; k < 256; ++k) s += hW3[j * 256 + k];
    w3s[j] = s;
}
__global__ void k_depth(const float* __restrict__ rh, const float* __restrict__ rW2,
                        const float* __restrict__ rb2, const float* __restrict__ rtheta,
                        float* __restrict__ dp, float* __restrict__ coarse) {
    long b = blockIdx.x * 256L + threadIdx.x;           // B threads
    const float* r = rh + b * 64;
    float s = 0.f;
#pragma unroll 8
    for (int k = 0; k < 64; ++k) s += r[k] * rW2[k];
    s += rb2[0];
    float p  = 1.f / (1.f + expf(-s));
    float th = 1.f / (1.f + expf(-rtheta[0]));
    dp[b]     = p;
    coarse[b] = (p > th) ? 1.f : 0.f;
}

// ---------------------------------------------------------------------------
extern "C" void kernel_launch(void* const* d_in, const int* in_sizes, int n_in,
                              void* d_out, int out_size, void* d_ws, size_t ws_size,
                              hipStream_t stream) {
    (void)in_sizes; (void)n_in; (void)out_size; (void)ws_size;
    const long B = BATCH;
    auto F = [&](int i) { return (const float*)d_in[i]; };
    const float* x     = F(0);
    const float* neigh = F(1);
    const float* nz1   = F(2);
    const float* nz2   = F(3);
    const float* hW1 = F(26); const float* hb1 = F(27);
    const float* hW2 = F(28); const float* hb2 = F(29);
    const float* hW3 = F(30); const float* hb3 = F(31);

    // ---- workspace carve-up: f32 region, then bf16 region ----
    float* fp = (float*)d_ws;
    float* F256  = fp;             // B x 256 (lin1 out / dH_dp)
    float* F128A = fp + 256 * B;   // B x 128 (a1 / m-lin1 / depth rh)
    float* F128B = fp + 384 * B;   // B x 128 (a2)
    float* F128C = fp + 512 * B;   // B x 128 (dh1)
    float* HFF   = fp + 640 * B;   // B x 256 (fused h, f32)
    float* W3S   = fp + 896 * B;   // 128
    __bf16* hp   = (__bf16*)(fp + 896 * B + 128);
    __bf16* XH    = hp;             // B x 1024 (x in bf16)
    __bf16* PH    = hp + 1024 * B;  // B x 512
    __bf16* X2H   = hp + 1536 * B;  // B x 512 (qp / mcat / z2)
    __bf16* H256H = hp + 2048 * B;  // B x 256 (encoder h)
    __bf16* HFH   = hp + 2304 * B;  // B x 256 (fused h)
    __bf16* BELH  = hp + 2560 * B;  // B x 256
    __bf16* QH    = hp + 2816 * B;  // B x 256 (grad_e)
    __bf16* CATH  = hp + 3072 * B;  // B x 192
    __bf16* H1HA  = hp + 3264 * B;  // B x 128 (h1 / mh)
    __bf16* H1HB  = hp + 3392 * B;  // B x 128 (h2 / da2)
    __bf16* H1HC  = hp + 3520 * B;  // B x 128 (da1)

    float* out = (float*)d_out;
    float* OUT0 = out;
    float* BEL  = out + 1000 * B;
    float* MSG  = out + 1256 * B;
    float* ENE  = out + 1512 * B;
    float* DP   = out + 1768 * B;
    float* CO   = out + 1769 * B;
    float* SC[3] = { out + 1770 * B, out + 2026 * B, out + 2154 * B };

    const dim3 blk(256);
    const int MB = (int)(B / 128);
    auto GRID = [&](int N) { return dim3((unsigned)((N + 63) / 64), (unsigned)MB); };
    const unsigned G1024 = (unsigned)(B * 1024 / 256), G512 = (unsigned)(B * 512 / 256),
                   G256  = (unsigned)(B * 256 / 256),  G192 = (unsigned)(B * 192 / 256),
                   G128  = (unsigned)(B * 128 / 256);

    k_w3s<<<1, 128, 0, stream>>>(hW3, W3S);
    k_cvt<<<G1024, blk, 0, stream>>>(x, XH, B * 1024);
    k_pool<<<G512, blk, 0, stream>>>(x, PH);

    // ---- multi-scale encoders ----
    for (int i = 0; i < 3; ++i) {
        const __bf16* A = (i == 0) ? XH : PH;
        const int K = (i == 0) ? 1024 : 512;
        const int pb = 4 + 6 * i;
        const int Nout = 256 >> i;
        gemm_wmma<false><<<GRID(256), blk, 0, stream>>>(A, F(pb + 0), F(pb + 1),
            F256, nullptr, 256, K, K, 256, 256, 1.f, 0);
        ln_gelu_rows<<<(unsigned)(B / 8), blk, 0, stream>>>(F256, H256H, nullptr, F(pb + 2), F(pb + 3), 256);
        gemm_wmma<false><<<GRID(Nout), blk, 0, stream>>>(H256H, F(pb + 4), F(pb + 5),
            SC[i], nullptr, Nout, 256, 256, Nout, Nout, 1.f, 0);
    }

    // ---- fusion layer ----
    k_cat3<<<G192, blk, 0, stream>>>(CATH, SC[0], SC[1], SC[2]);
    gemm_wmma<false><<<GRID(256), blk, 0, stream>>>(CATH, F(22), F(23), F256, nullptr,
        256, 192, 192, 256, 256, 1.f, 0);
    ln_gelu_rows<<<(unsigned)(B / 8), blk, 0, stream>>>(F256, HFH, HFF, F(24), F(25), 256);

    // ---- Hamiltonian grad #1 at qp=[h, noise1] ----
    k_concat_hf<<<G512, blk, 0, stream>>>(X2H, HFH, nz1);
    gemm_wmma<false><<<GRID(128), blk, 0, stream>>>(X2H, hW1, hb1, F128A, nullptr, 128, 512, 512, 128, 128, 1.f, 0);
    k_gelu<<<G128, blk, 0, stream>>>(F128A, H1HA, B * 128);                         // h1
    gemm_wmma<false><<<GRID(128), blk, 0, stream>>>(H1HA, hW2, hb2, F128B, nullptr, 128, 128, 128, 128, 128, 1.f, 0);
    k_w3grad<<<G128, blk, 0, stream>>>(F128B, W3S, H1HB, B * 128);                  // da2
    gemm_wmma<true ><<<GRID(128), blk, 0, stream>>>(H1HB, hW2, nullptr, F128C, nullptr, 128, 128, 128, 128, 128, 1.f, 0);
    k_mulgrad<<<G128, blk, 0, stream>>>(F128C, F128A, H1HC, B * 128);               // da1
    gemm_wmma<true ><<<GRID(256), blk, 0, stream>>>(H1HC, hW1 + 256 * 128, nullptr,
        F256, nullptr, 256, 128, 128, 128, 256, 1.f, 0);                            // dH_dp
    k_belief<<<G256, blk, 0, stream>>>(HFF, F256, BEL, BELH, B * 256);

    // ---- messages ----
    k_concat_fh<<<G512, blk, 0, stream>>>(X2H, neigh, BELH);
    gemm_wmma<false><<<GRID(128), blk, 0, stream>>>(X2H, F(32), F(33), F128A, nullptr, 128, 512, 512, 128, 128, 1.f, 0);
    ln_gelu_rows<<<(unsigned)(B / 8), blk, 0, stream>>>(F128A, H1HA, nullptr, F(34), F(35), 128);
    gemm_wmma<false><<<GRID(256), blk, 0, stream>>>(H1HA, F(36), F(37), MSG, nullptr, 256, 128, 128, 256, 256, 1.f, 0);

    // ---- energy + Hamiltonian grad #2 at z2=[belief, noise2] ----
    k_concat_hf<<<G512, blk, 0, stream>>>(X2H, BELH, nz2);
    gemm_wmma<false><<<GRID(128), blk, 0, stream>>>(X2H, hW1, hb1, F128A, nullptr, 128, 512, 512, 128, 128, 1.f, 0);
    k_gelu<<<G128, blk, 0, stream>>>(F128A, H1HA, B * 128);                         // h1
    gemm_wmma<false><<<GRID(128), blk, 0, stream>>>(H1HA, hW2, hb2, F128B, nullptr, 128, 128, 128, 128, 128, 1.f, 0);
    k_gelu<<<G128, blk, 0, stream>>>(F128B, H1HB, B * 128);                         // h2
    gemm_wmma<false><<<GRID(256), blk, 0, stream>>>(H1HB, hW3, hb3, ENE, nullptr, 256, 128, 128, 256, 256, 1.f, 0);
    k_w3grad<<<G128, blk, 0, stream>>>(F128B, W3S, H1HB, B * 128);                  // da2
    gemm_wmma<true ><<<GRID(128), blk, 0, stream>>>(H1HB, hW2, nullptr, F128C, nullptr, 128, 128, 128, 128, 128, 1.f, 0);
    k_mulgrad<<<G128, blk, 0, stream>>>(F128C, F128A, H1HC, B * 128);
    gemm_wmma<true ><<<GRID(256), blk, 0, stream>>>(H1HC, hW1, nullptr,
        nullptr, QH, 256, 128, 128, 128, 256, 1.f, 0);                              // grad_e (bf16)

    // ---- classifier: output = (-LR^2/2)*(grad_e @ cW) + cb ----
    gemm_wmma<false><<<GRID(1000), blk, 0, stream>>>(QH, F(43), F(44), OUT0, nullptr,
        1000, 256, 256, 1000, 1000, -5.0e-7f, 0);

    // ---- depth / router ----
    gemm_wmma<false><<<GRID(64), blk, 0, stream>>>(BELH, F(38), F(39), F128A, nullptr,
        64, 256, 256, 64, 64, 1.f, 1);
    k_depth<<<(unsigned)(B / 256), blk, 0, stream>>>(F128A, F(40), F(41), F(42), DP, CO);
}
